// HybridGraphRefiner_57019985822410
// MI455X (gfx1250) — matrix operand order repaired and checked by reading, not compiled, not run on hardware
//
#include <hip/hip_runtime.h>
#include <hip/hip_bf16.h>

typedef __attribute__((ext_vector_type(16))) _Float16 v16h;
typedef __attribute__((ext_vector_type(8)))  float    v8f;

#define CC    256
#define NN    4096
#define HWDIM 64
#define HIDD  128
#define NPP   16
#define NCC   16
#define BB    2

// ---------------- WMMA fragment helpers (gfx1250, wave32) ----------------
// A fragment: 16x32 f16 tile, rows m, cols k, tile row-major [m][k].
// Per ISA: lanes 0-15 -> K {0-7,16-23}, lanes 16-31 -> K {8-15,24-31};
// each kh-half is two contiguous 8-element runs -> 2x ds_load_b128.
__device__ __forceinline__ v16h frag_a(const _Float16* t, int m0, int k0, int ld) {
  const int lane = threadIdx.x & 31;
  const int kh = lane >> 4;
  const _Float16* p = t + (size_t)(m0 + (lane & 15)) * ld + k0;
  v16h a;
#pragma unroll
  for (int r = 0; r < 8; ++r) {
    int kb = (r < 4) ? (kh * 8 + 2 * r) : (16 + kh * 8 + 2 * (r - 4));
    a[2 * r]     = p[kb];
    a[2 * r + 1] = p[kb + 1];
  }
  return a;
}

// B fragment from a TRANSPOSED tile stored [n][k] (k contiguous, stride ld).
// Per-lane B elements are K {kh*8..kh*8+7, 16+kh*8..16+kh*8+7} at fixed n
// (matching frag_a) -> 2x ds_load_b128. ld must be even and rows
// 16B-aligned (ld multiple of 8 f16).
__device__ __forceinline__ v16h frag_b_t(const _Float16* t, int k0, int n0, int ld) {
  const int lane = threadIdx.x & 31;
  const int kh = lane >> 4;
  const _Float16* p = t + (size_t)(n0 + (lane & 15)) * ld + k0 + kh * 8;
  v16h b;
#pragma unroll
  for (int e = 0; e < 16; ++e) b[e] = p[e < 8 ? e : e + 8];
  return b;
}

__device__ __forceinline__ v8f wmma_f16(v16h a, v16h b, v8f c) {
  return __builtin_amdgcn_wmma_f32_16x16x32_f16(false, a, false, b, (short)0, c,
                                                false, false);
}

__device__ __forceinline__ void zero8(v8f& c) {
#pragma unroll
  for (int r = 0; r < 8; ++r) c[r] = 0.f;
}

// ---------------- Kernel 1: fused dual LayerNorm + f16 pack ----------------
__global__ void __launch_bounds__(256) ln_pack(const float* __restrict__ x,
                                               const float* __restrict__ g1,
                                               const float* __restrict__ b1,
                                               const float* __restrict__ g2,
                                               const float* __restrict__ b2,
                                               _Float16* __restrict__ x1,
                                               _Float16* __restrict__ x2) {
  const int b = blockIdx.x >> 12;      // NN = 4096
  const int n = blockIdx.x & (NN - 1);
  const int c = threadIdx.x;           // CC = 256 threads
  __shared__ float red[256];
  const size_t idx = ((size_t)b * CC + c) * NN + n;
  float v = x[idx];
  red[c] = v;
  __syncthreads();
  for (int s = 128; s > 0; s >>= 1) {
    if (c < s) red[c] += red[c + s];
    __syncthreads();
  }
  const float mean = red[0] * (1.f / CC);
  __syncthreads();
  const float d = v - mean;
  red[c] = d * d;
  __syncthreads();
  for (int s = 128; s > 0; s >>= 1) {
    if (c < s) red[c] += red[c + s];
    __syncthreads();
  }
  const float inv = rsqrtf(red[0] * (1.f / CC) + 1e-5f);
  const float xn = d * inv;
  x1[idx] = (_Float16)(xn * g1[c] + b1[c]);
  x2[idx] = (_Float16)(xn * g2[c] + b2[c]);
}

// ------------- Kernel 2: archetype gate -> per-row scale vector -------------
__global__ void __launch_bounds__(256) gate_rs(const float* __restrict__ arch,
                                               const float* __restrict__ Wg,
                                               const float* __restrict__ bg,
                                               float* __restrict__ rs) {
  const int b = blockIdx.x;
  const int c = threadIdx.x;
  __shared__ float guide[CC];
  float s = 0.f;
  for (int p = 0; p < NPP; ++p) s += arch[((size_t)b * NPP + p) * CC + c];
  guide[c] = s * (1.f / NPP);
  __syncthreads();
  float acc = bg[c];
  for (int k = 0; k < CC; ++k) acc += Wg[(size_t)c * CC + k] * guide[k];
  const float g = 1.f / (1.f + __expf(-acc));
  rs[(size_t)b * 768 + c] = 0.0625f;        // q: softmax scale 1/sqrt(256)
  rs[(size_t)b * 768 + 256 + c] = g;        // k: archetype gate
  rs[(size_t)b * 768 + 512 + c] = 1.0f;     // v
}

// ---------------- Kernel 3: contextual KV projection (tiny) ----------------
__global__ void __launch_bounds__(256) kv_proj(const float* __restrict__ ctx,
                                               const float* __restrict__ Wkv,
                                               const float* __restrict__ bkv,
                                               float* __restrict__ kc,
                                               float* __restrict__ vc) {
  const int b = blockIdx.x;
  const int o = threadIdx.x;  // 0..255 = 2*HID rows
  for (int j = 0; j < NCC; ++j) {
    float acc = bkv[o];
    const float* cp = ctx + ((size_t)b * NCC + j) * CC;
    for (int k = 0; k < CC; ++k) acc += Wkv[(size_t)o * CC + k] * cp[k];
    if (o < HIDD)
      kc[((size_t)b * HIDD + o) * NCC + j] = acc;
    else
      vc[((size_t)b * HIDD + (o - HIDD)) * NCC + j] = acc;
  }
}

// ---------------- Generic WMMA GEMM: Y[b] = W @ X[b] (+bias etc.) ----------------
// MODE 0: f16 out.  MODE 2: +resid, f16 out.  MODE 3: +resid, f32 out.
template <int MODE>
__global__ void __launch_bounds__(256) gemm_wx(
    const float* __restrict__ W, const _Float16* __restrict__ X,
    const float* __restrict__ bias, const float* __restrict__ rowscale, long rsb,
    float uscale, _Float16* __restrict__ Yh, float* __restrict__ Yf,
    const float* __restrict__ resid, int M, int K, long xb, long yb, long rb) {
  __shared__ _Float16 as[64 * 40];    // 64 m x 32 k (pad 8)
  __shared__ _Float16 bsT[128 * 40];  // TRANSPOSED: 128 n x 32 k (pad 8)
  const int tid = threadIdx.x;
  const int wid = tid >> 5;
  const int lane = tid & 31;
  const int b = blockIdx.z;
  const int m0 = blockIdx.y * 64;
  const int n0 = blockIdx.x * 128;
  const int m0w = (wid >> 2) * 32;
  const int n0w = (wid & 3) * 32;
  const _Float16* Xb = X + (size_t)b * xb;

  v8f acc[2][2];
#pragma unroll
  for (int mi = 0; mi < 2; ++mi)
#pragma unroll
    for (int ni = 0; ni < 2; ++ni) zero8(acc[mi][ni]);

  for (int k0 = 0; k0 < K; k0 += 32) {
    __syncthreads();
    for (int i = tid; i < 64 * 32; i += 256) {
      int r = i >> 5, c = i & 31;
      as[r * 40 + c] = (_Float16)W[(size_t)(m0 + r) * K + k0 + c];
    }
    for (int i = tid; i < 32 * 128; i += 256) {
      int r = i >> 7, c = i & 127;  // consecutive tid -> consecutive n (coalesced)
      bsT[c * 40 + r] = Xb[(size_t)(k0 + r) * NN + n0 + c];
    }
    if (k0 + 32 < K) {  // prefetch next K-slice (global_prefetch_b8)
      __builtin_prefetch(Xb + (size_t)(k0 + 32) * NN + n0 + tid, 0, 1);
      __builtin_prefetch(W + (size_t)(m0 + (tid & 63)) * K + k0 + 32, 0, 1);
    }
    __syncthreads();
    v16h af[2], bf[2];
#pragma unroll
    for (int mi = 0; mi < 2; ++mi) af[mi] = frag_a(as, m0w + 16 * mi, 0, 40);
#pragma unroll
    for (int ni = 0; ni < 2; ++ni) bf[ni] = frag_b_t(bsT, 0, n0w + 16 * ni, 40);
#pragma unroll
    for (int mi = 0; mi < 2; ++mi)
#pragma unroll
      for (int ni = 0; ni < 2; ++ni) acc[mi][ni] = wmma_f16(af[mi], bf[ni], acc[mi][ni]);
  }

#pragma unroll
  for (int mi = 0; mi < 2; ++mi)
#pragma unroll
    for (int ni = 0; ni < 2; ++ni)
#pragma unroll
      for (int r = 0; r < 8; ++r) {
        const int row = m0 + m0w + 16 * mi + r + 8 * (lane >> 4);
        const int col = n0 + n0w + 16 * ni + (lane & 15);
        float v = acc[mi][ni][r];
        if (bias) v += bias[row];
        if (rowscale) v *= rowscale[(size_t)b * rsb + row];
        v *= uscale;
        if (MODE >= 2) v += resid[(size_t)b * rb + (size_t)row * NN + col];
        const size_t o = (size_t)b * yb + (size_t)row * NN + col;
        if (MODE == 3)
          Yf[o] = v;
        else
          Yh[o] = (_Float16)v;
      }
}

// ---------------- Flash self-attention: seq 4096, dim 256, WMMA ----------------
// qkv f16 [B,768,N] (q pre-scaled 1/16, k pre-gated). Writes ref_s f16 [B,256,N].
__global__ void __launch_bounds__(256) flash_self(const _Float16* __restrict__ qkv,
                                                  _Float16* __restrict__ refs) {
  extern __shared__ char smem[];
  _Float16* qs = (_Float16*)smem;           // [64][256]   Q^T (A tile, row-major k)
  _Float16* vt = qs + 64 * 256;             // [256][72]   V slice [c][m] (B^T tile)
  float* sS = (float*)(vt + 256 * 72);      // [64][68]    S scores
  _Float16* pP = (_Float16*)(sS + 64 * 68); // [64][72]    P = exp(S - m)
  _Float16* kt = pP + 64 * 72;              // [64][40]    K slice [j][c-slice] (B^T)
  float* mst = (float*)(kt + 64 * 40);      // [64] running max
  float* lst = mst + 64;                    // [64] running sum
  float* ast = lst + 64;                    // [64] alpha

  const int tid = threadIdx.x;
  const int wid = tid >> 5;
  const int lane = tid & 31;
  const int b = blockIdx.y;
  const int q0 = blockIdx.x * 64;
  const _Float16* qp = qkv + (size_t)b * 768 * NN;
  const _Float16* kp = qp + (size_t)256 * NN;
  const _Float16* vp = qp + (size_t)512 * NN;

  for (int i = tid; i < 64 * 256; i += 256) {
    const int qi = i & 63, c = i >> 6;   // coalesced global reads along qi
    qs[qi * 256 + c] = qp[(size_t)c * NN + q0 + qi];
  }
  if (tid < 64) { mst[tid] = -1e30f; lst[tid] = 0.f; }

  const int sq0 = (wid & 1) * 32;   // S wave rows
  const int sj0 = (wid >> 1) * 16;  // S wave cols
  const int oq0 = (wid >> 2) * 32;  // O wave rows
  const int oc0 = (wid & 3) * 64;   // O wave cols

  v8f oacc[2][4];
#pragma unroll
  for (int mi = 0; mi < 2; ++mi)
#pragma unroll
    for (int ni = 0; ni < 4; ++ni) zero8(oacc[mi][ni]);
  __syncthreads();

  for (int j0 = 0; j0 < NN; j0 += 64) {
    __syncthreads();  // prior O-GEMM done reading vt/pP
    {  // V slice: vt[c][m] = vp[c][j0+m]; contiguous both sides -> uint4 copy
      const uint4* src = (const uint4*)(vp + (size_t)tid * NN + j0);
      uint4* dst = (uint4*)(vt + (size_t)tid * 72);
#pragma unroll
      for (int u = 0; u < 8; ++u) dst[u] = src[u];
    }
    v8f sacc[2];
    zero8(sacc[0]);
    zero8(sacc[1]);
    for (int kk = 0; kk < 256; kk += 32) {
      __syncthreads();
      for (int i = tid; i < 32 * 64; i += 256) {
        const int kr = i >> 6, j = i & 63;  // coalesced reads along j
        kt[j * 40 + kr] = kp[(size_t)(kk + kr) * NN + j0 + j];
      }
      __syncthreads();
      const v16h bk = frag_b_t(kt, 0, sj0, 40);
#pragma unroll
      for (int mi = 0; mi < 2; ++mi) {
        const v16h aq = frag_a(qs, sq0 + 16 * mi, kk, 256);
        sacc[mi] = wmma_f16(aq, bk, sacc[mi]);
      }
    }
#pragma unroll
    for (int mi = 0; mi < 2; ++mi)
#pragma unroll
      for (int r = 0; r < 8; ++r) {
        const int row = sq0 + 16 * mi + r + 8 * (lane >> 4);
        sS[row * 68 + sj0 + (lane & 15)] = sacc[mi][r];
      }
    __syncthreads();
    if (tid < 64) {  // online softmax, one row per thread
      float* rp = sS + tid * 68;
      float mx = mst[tid];
      for (int j = 0; j < 64; ++j) mx = fmaxf(mx, rp[j]);
      const float alpha = __expf(mst[tid] - mx);
      float l = lst[tid] * alpha;
      _Float16* pr = pP + tid * 72;
      for (int j = 0; j < 64; ++j) {
        const float e = __expf(rp[j] - mx);
        l += e;
        pr[j] = (_Float16)e;
      }
      mst[tid] = mx;
      lst[tid] = l;
      ast[tid] = alpha;
    }
    __syncthreads();
#pragma unroll
    for (int mi = 0; mi < 2; ++mi)
#pragma unroll
      for (int r = 0; r < 8; ++r) {
        const float al = ast[oq0 + 16 * mi + r + 8 * (lane >> 4)];
#pragma unroll
        for (int ni = 0; ni < 4; ++ni) oacc[mi][ni][r] *= al;
      }
#pragma unroll
    for (int mk = 0; mk < 64; mk += 32) {
      v16h ap[2], bv[4];
#pragma unroll
      for (int mi = 0; mi < 2; ++mi) ap[mi] = frag_a(pP, oq0 + 16 * mi, mk, 72);
#pragma unroll
      for (int ni = 0; ni < 4; ++ni) bv[ni] = frag_b_t(vt, mk, oc0 + 16 * ni, 72);
#pragma unroll
      for (int mi = 0; mi < 2; ++mi)
#pragma unroll
        for (int ni = 0; ni < 4; ++ni) oacc[mi][ni] = wmma_f16(ap[mi], bv[ni], oacc[mi][ni]);
    }
  }
  _Float16* ob = refs + (size_t)b * 256 * NN;
#pragma unroll
  for (int mi = 0; mi < 2; ++mi)
#pragma unroll
    for (int r = 0; r < 8; ++r) {
      const int row = oq0 + 16 * mi + r + 8 * (lane >> 4);
      const float inv = 1.f / lst[row];
#pragma unroll
      for (int ni = 0; ni < 4; ++ni) {
        const int c = oc0 + 16 * ni + (lane & 15);
        ob[(size_t)c * NN + q0 + row] = (_Float16)(oacc[mi][ni][r] * inv);
      }
    }
}

// ---------------- Cross-attention over 16 contextual tokens ----------------
__global__ void __launch_bounds__(256) cross_attn(const _Float16* __restrict__ qc,
                                                  const float* __restrict__ kc,
                                                  const float* __restrict__ vc,
                                                  _Float16* __restrict__ refc) {
  const int b = blockIdx.y;
  const int n = blockIdx.x * 256 + threadIdx.x;
  __shared__ float kcl[HIDD * NCC];
  __shared__ float vcl[HIDD * NCC];
  for (int i = threadIdx.x; i < HIDD * NCC; i += 256) {
    kcl[i] = kc[(size_t)b * HIDD * NCC + i];
    vcl[i] = vc[(size_t)b * HIDD * NCC + i];
  }
  __syncthreads();
  float acc[NCC];
#pragma unroll
  for (int j = 0; j < NCC; ++j) acc[j] = 0.f;
  const _Float16* qb = qc + (size_t)b * HIDD * NN + n;
  for (int h = 0; h < HIDD; ++h) {
    const float qv = (float)qb[(size_t)h * NN];
#pragma unroll
    for (int j = 0; j < NCC; ++j) acc[j] += qv * kcl[h * NCC + j];
  }
  float mx = acc[0];
#pragma unroll
  for (int j = 1; j < NCC; ++j) mx = fmaxf(mx, acc[j]);
  float s = 0.f;
#pragma unroll
  for (int j = 0; j < NCC; ++j) {
    acc[j] = __expf(acc[j] - mx);
    s += acc[j];
  }
  const float inv = 1.f / s;
  _Float16* ob = refc + (size_t)b * HIDD * NN + n;
  for (int h = 0; h < HIDD; ++h) {
    float o = 0.f;
#pragma unroll
    for (int j = 0; j < NCC; ++j) o += vcl[h * NCC + j] * acc[j];
    ob[(size_t)h * NN] = (_Float16)(o * inv);
  }
}

// ---------------- 3x3 conv (512->256 ch) as implicit WMMA GEMM + ReLU ----------------
__global__ void __launch_bounds__(256) conv3x3_relu(const float* __restrict__ Wf1,
                                                    const _Float16* __restrict__ fused,
                                                    _Float16* __restrict__ out) {
  __shared__ _Float16 as[64 * 40];   // 64 o x 32 i (pad 8)
  __shared__ _Float16 bsT[64 * 40];  // TRANSPOSED: 64 x-positions x 32 i (pad 8)
  const int tid = threadIdx.x;
  const int wid = tid >> 5;
  const int lane = tid & 31;
  const int y = blockIdx.x;        // row 0..63
  const int o0 = blockIdx.y * 64;  // output-channel block
  const int b = blockIdx.z;
  const int m0w = (wid & 1) * 32;
  const int n0w = (wid >> 1) * 16;
  const _Float16* fb = fused + (size_t)b * 512 * NN;

  v8f acc[2];
  zero8(acc[0]);
  zero8(acc[1]);

  for (int dy = -1; dy <= 1; ++dy) {
    const int yy = y + dy;
    const bool yok = (yy >= 0) && (yy < HWDIM);
    for (int dx = -1; dx <= 1; ++dx) {
      for (int i0 = 0; i0 < 512; i0 += 32) {
        __syncthreads();
        for (int t = tid; t < 64 * 32; t += 256) {
          const int r = t >> 5, c = t & 31;
          as[r * 40 + c] = (_Float16)
              Wf1[(((size_t)(o0 + r) * 512 + (i0 + c)) * 3 + (dy + 1)) * 3 + (dx + 1)];
        }
        for (int t = tid; t < 32 * 64; t += 256) {
          const int r = t >> 6, xx = t & 63;  // coalesced reads along x
          const int xs = xx + dx;
          _Float16 v = (_Float16)0.f;
          if (yok && xs >= 0 && xs < HWDIM)
            v = fb[(size_t)(i0 + r) * NN + yy * HWDIM + xs];
          bsT[xx * 40 + r] = v;
        }
        __syncthreads();
        const v16h bf = frag_b_t(bsT, 0, n0w, 40);
#pragma unroll
        for (int mi = 0; mi < 2; ++mi) {
          const v16h af = frag_a(as, m0w + 16 * mi, 0, 40);
          acc[mi] = wmma_f16(af, bf, acc[mi]);
        }
      }
    }
  }
  _Float16* ob = out + (size_t)b * CC * NN;
#pragma unroll
  for (int mi = 0; mi < 2; ++mi)
#pragma unroll
    for (int r = 0; r < 8; ++r) {
      const int row = o0 + m0w + 16 * mi + r + 8 * (lane >> 4);
      const int col = n0w + (lane & 15);
      const float v = fmaxf(acc[mi][r], 0.f);
      ob[(size_t)row * NN + y * HWDIM + col] = (_Float16)v;
    }
}

// ---------------- host-side orchestration ----------------
extern "C" void kernel_launch(void* const* d_in, const int* in_sizes, int n_in,
                              void* d_out, int out_size, void* d_ws, size_t ws_size,
                              hipStream_t stream) {
  (void)in_sizes; (void)n_in; (void)out_size; (void)ws_size;
  const float* qry   = (const float*)d_in[0];
  const float* arch  = (const float*)d_in[1];
  const float* ctx   = (const float*)d_in[2];
  const float* ln1g  = (const float*)d_in[3];
  const float* ln1b  = (const float*)d_in[4];
  const float* Wqkv  = (const float*)d_in[5];
  const float* bqkv  = (const float*)d_in[6];
  const float* Wproj = (const float*)d_in[7];
  const float* bproj = (const float*)d_in[8];
  const float* Wg    = (const float*)d_in[9];
  const float* bg    = (const float*)d_in[10];
  const float* ln2g  = (const float*)d_in[11];
  const float* ln2b  = (const float*)d_in[12];
  const float* Wq    = (const float*)d_in[13];
  const float* bq    = (const float*)d_in[14];
  const float* Wkv   = (const float*)d_in[15];
  const float* bkv   = (const float*)d_in[16];
  const float* Wout  = (const float*)d_in[17];
  const float* bout  = (const float*)d_in[18];
  const float* Wf1   = (const float*)d_in[19];
  const float* Wf2   = (const float*)d_in[20];
  const float* bf2   = (const float*)d_in[21];
  float* outp = (float*)d_out;

  char* ws = (char*)d_ws;
  size_t off = 0;
  auto alloc = [&](size_t bytes) -> char* {
    char* p = ws + off;
    off += (bytes + 255) & ~(size_t)255;
    return p;
  };
  _Float16* x1    = (_Float16*)alloc((size_t)BB * CC * NN * 2);
  _Float16* x2    = (_Float16*)alloc((size_t)BB * CC * NN * 2);
  _Float16* qkvb  = (_Float16*)alloc((size_t)BB * 768 * NN * 2);
  _Float16* refs  = (_Float16*)alloc((size_t)BB * CC * NN * 2);
  _Float16* qc    = (_Float16*)alloc((size_t)BB * HIDD * NN * 2);
  _Float16* refc  = (_Float16*)alloc((size_t)BB * HIDD * NN * 2);
  _Float16* fusedb= (_Float16*)alloc((size_t)BB * 512 * NN * 2);
  _Float16* c1    = (_Float16*)alloc((size_t)BB * CC * NN * 2);
  float* kc = (float*)alloc((size_t)BB * HIDD * NCC * 4);
  float* vc = (float*)alloc((size_t)BB * HIDD * NCC * 4);
  float* rs = (float*)alloc((size_t)BB * 768 * 4);

  const dim3 blk(256);

  // 1) LayerNorms
  ln_pack<<<dim3(BB * NN), blk, 0, stream>>>(qry, ln1g, ln1b, ln2g, ln2b, x1, x2);
  // 2) gate -> rowscale ; contextual kv
  gate_rs<<<dim3(BB), blk, 0, stream>>>(arch, Wg, bg, rs);
  kv_proj<<<dim3(BB), blk, 0, stream>>>(ctx, Wkv, bkv, kc, vc);
  // 3) QKV projection (q scaled 1/16, k gated, via rowscale)
  gemm_wx<0><<<dim3(NN / 128, 768 / 64, BB), blk, 0, stream>>>(
      Wqkv, x1, bqkv, rs, 768, 1.0f, qkvb, nullptr, nullptr,
      768, 256, (long)CC * NN, (long)768 * NN, 0);
  // 4) flash self-attention
  const size_t fsm = (size_t)64 * 256 * 2 + 256 * 72 * 2 + 64 * 68 * 4 +
                     64 * 72 * 2 + 64 * 40 * 2 + 3 * 64 * 4;  // 102,144 B
  flash_self<<<dim3(NN / 64, BB), blk, fsm, stream>>>(qkvb, refs);
  // 5) Wproj + residual -> fused[0:256]
  gemm_wx<2><<<dim3(NN / 128, 4, BB), blk, 0, stream>>>(
      Wproj, refs, bproj, nullptr, 0, 1.0f, fusedb, nullptr, qry,
      256, 256, (long)CC * NN, (long)512 * NN, (long)CC * NN);
  // 6) Wq projection (scale 1/sqrt(128) folded)
  gemm_wx<0><<<dim3(NN / 128, 2, BB), blk, 0, stream>>>(
      Wq, x2, bq, nullptr, 0, 0.08838834764831845f, qc, nullptr, nullptr,
      128, 256, (long)CC * NN, (long)HIDD * NN, 0);
  // 7) cross attention over 16 tokens
  cross_attn<<<dim3(NN / 256, BB), blk, 0, stream>>>(qc, kc, vc, refc);
  // 8) Wout + residual -> fused[256:512]
  gemm_wx<2><<<dim3(NN / 128, 4, BB), blk, 0, stream>>>(
      Wout, refc, bout, nullptr, 0, 1.0f, fusedb + (size_t)256 * NN, nullptr, qry,
      256, 128, (long)HIDD * NN, (long)512 * NN, (long)CC * NN);
  // 9) 3x3 conv + ReLU
  conv3x3_relu<<<dim3(HWDIM, 4, BB), blk, 0, stream>>>(Wf1, fusedb, c1);
  // 10) 1x1 conv + bf2 + qry residual -> f32 output
  gemm_wx<3><<<dim3(NN / 128, 4, BB), blk, 0, stream>>>(
      Wf2, c1, bf2, nullptr, 0, 1.0f, nullptr, outp, qry,
      256, 256, (long)CC * NN, (long)CC * NN, (long)CC * NN);
}
